// mLSTM_block_87883620811268
// MI455X (gfx1250) — matrix operand order repaired and checked by reading, not compiled, not run on hardware
//
#include <hip/hip_runtime.h>

// MI455X / gfx1250: wave32, WMMA 16x16x32 bf16 -> f32.
typedef __attribute__((ext_vector_type(16))) __bf16 v16bf;
typedef __attribute__((ext_vector_type(8)))  float  v8f;

#define WMMA_BF16(a, b, c) \
  __builtin_amdgcn_wmma_f32_16x16x32_bf16(false, (a), false, (b), (short)0, (c), false, false)

static __device__ __forceinline__ float silu_f(float v) {
  return v / (1.0f + __expf(-v));
}

// Async global->LDS copy of 16B (per-lane), ASYNCcnt-tracked on gfx1250.
// Inline asm (per CDNA5 ISA): vdst = LDS byte offset VGPR,
// vaddr = 64-bit global address.
static __device__ __forceinline__ void async_copy_b128(const __bf16* g, __bf16* l) {
  unsigned loff = (unsigned)(size_t)(__attribute__((address_space(3))) __bf16*)l;
  asm volatile("global_load_async_to_lds_b128 %0, %1, off"
               :: "v"(loff), "v"(g) : "memory");
}
static __device__ __forceinline__ void async_copy_wait() {
  asm volatile("s_wait_asynccnt 0" ::: "memory");
}

// ---------------------------------------------------------------------------
// Zero the 3 causal-conv pad rows per batch of a_pad [2][2051][1024] bf16.
// ---------------------------------------------------------------------------
__global__ void k_zero_pad(__bf16* __restrict__ a_pad) {
  int t = blockIdx.x * 256 + threadIdx.x;
  if (t < 2 * 3 * 1024) {
    int b = t / 3072, off = t % 3072;
    a_pad[(size_t)b * 2051 * 1024 + off] = (__bf16)0.0f;
  }
}

// ---------------------------------------------------------------------------
// LayerNorm over DIM=512, output bf16. One block per row, 256 threads.
// ---------------------------------------------------------------------------
__global__ void k_layernorm(const float* __restrict__ x, const float* __restrict__ w,
                            const float* __restrict__ b, __bf16* __restrict__ out) {
  int row = blockIdx.x, tid = threadIdx.x;
  const float* xr = x + (size_t)row * 512;
  float v0 = xr[tid], v1 = xr[tid + 256];
  __shared__ float red0[256], red1[256];
  red0[tid] = v0 + v1;
  red1[tid] = v0 * v0 + v1 * v1;
  __syncthreads();
  for (int o = 128; o > 0; o >>= 1) {
    if (tid < o) { red0[tid] += red0[tid + o]; red1[tid] += red1[tid + o]; }
    __syncthreads();
  }
  float mean = red0[0] * (1.0f / 512.0f);
  float var  = red1[0] * (1.0f / 512.0f) - mean * mean;
  float rstd = rsqrtf(var + 1e-5f);
  out[(size_t)row * 512 + tid]       = (__bf16)((v0 - mean) * rstd * w[tid] + b[tid]);
  out[(size_t)row * 512 + tid + 256] = (__bf16)((v1 - mean) * rstd * w[tid + 256] + b[tid + 256]);
}

// ---------------------------------------------------------------------------
// Weight convert+transpose: dst[n*K+k] = (bf16)src[k*sk + n*sn].
// ---------------------------------------------------------------------------
__global__ void k_w2bf16t(const float* __restrict__ src, __bf16* __restrict__ dst,
                          int K, int N, int sk, int sn) {
  int t = blockIdx.x * 256 + threadIdx.x;
  if (t >= K * N) return;
  int k = t % K, n = t / K;
  dst[(size_t)n * K + k] = (__bf16)src[(size_t)k * sk + (size_t)n * sn];
}

// ---------------------------------------------------------------------------
// WMMA GEMM, 128x128 tile, 256 threads (8 waves), each wave 32x64 (2x4 frags).
// C[M,N] = act(sum_kc A[m+kc, :] * Btk^T + bias (+ addsrc))
//   A   : bf16 [.., lda] row-major (for ksteps=4 rows are the shifted conv taps)
//   Bt  : bf16, ksteps matrices of [N, K] row-major (transposed weights)
// B-tile (128 x 32) double-buffered in LDS via async global->LDS copies:
// the copy for step i+1 is in flight during all 8 WMMAs of step i, with a
// single workgroup barrier per k-step.
// ---------------------------------------------------------------------------
__global__ void k_gemm(const __bf16* __restrict__ A, int lda,
                       const __bf16* __restrict__ Bt,
                       const float* __restrict__ bias,
                       const float* __restrict__ addsrc,
                       float* __restrict__ Cf, __bf16* __restrict__ Cbf, int ldc,
                       int K, int N, int ksteps, int act) {
  int tid = threadIdx.x;
  int lane = tid & 31, w = tid >> 5;
  int m0 = blockIdx.y * 128 + (w >> 1) * 32;   // wave's 32 rows
  int nb = blockIdx.x * 128;                   // block's 128 cols
  int nw = (w & 1) * 64;                       // wave's 64-col half
  int lr = lane & 15, lh = lane >> 4;

  __shared__ alignas(64) __bf16 Bst[2][128][48];  // 32 k-values + pad, x2 buffers

  int stg_n = tid >> 1;          // 0..127 : which C-column row of Bst
  int stg_h = (tid & 1) * 16;    // 0/16   : which 16-k half

  // Pre-issue the step-0 copy into buffer 0.
  async_copy_b128(Bt + (size_t)(nb + stg_n) * K + stg_h, &Bst[0][stg_n][stg_h]);
  async_copy_b128(Bt + (size_t)(nb + stg_n) * K + stg_h + 8, &Bst[0][stg_n][stg_h + 8]);

  v8f acc[2][4] = {};
  int buf = 0;
  for (int kc = 0; kc < ksteps; kc++) {
    const __bf16* Ak = A + (size_t)kc * lda;             // shifted rows (conv taps)
    for (int kk = 0; kk < K; kk += 32) {
      // My copy for this step (issued one compute-phase ago) is complete...
      async_copy_wait();
      // ...and after this barrier, everyone's is, and everyone has finished
      // reading the buffer we are about to overwrite.
      __syncthreads();

      // Issue next step's copy into the other buffer (overlaps the WMMAs).
      int nkk = kk + 32, nkc = kc;
      if (nkk == K) { nkk = 0; nkc = kc + 1; }
      if (nkc < ksteps) {
        async_copy_b128(Bt + (size_t)nkc * N * K + (size_t)(nb + stg_n) * K + nkk + stg_h,
                        &Bst[buf ^ 1][stg_n][stg_h]);
        async_copy_b128(Bt + (size_t)nkc * N * K + (size_t)(nb + stg_n) * K + nkk + stg_h + 8,
                        &Bst[buf ^ 1][stg_n][stg_h + 8]);
      }

      v16bf a0 = *(const v16bf*)(Ak + (size_t)(m0 + lr) * lda      + kk + lh * 16);
      v16bf a1 = *(const v16bf*)(Ak + (size_t)(m0 + 16 + lr) * lda + kk + lh * 16);
      for (int f = 0; f < 4; f++) {
        v16bf bv = *(const v16bf*)(&Bst[buf][nw + f * 16 + lr][lh * 16]);
        acc[0][f] = WMMA_BF16(a0, bv, acc[0][f]);
        acc[1][f] = WMMA_BF16(a1, bv, acc[1][f]);
      }
      buf ^= 1;
    }
  }
  for (int s = 0; s < 2; s++) {
    for (int f = 0; f < 4; f++) {
      int n = nb + nw + f * 16 + lr;
      float bv = bias ? bias[n] : 0.0f;
      for (int r = 0; r < 8; r++) {
        int m = m0 + s * 16 + r + lh * 8;
        float v = acc[s][f][r] + bv;
        if (addsrc) v += addsrc[(size_t)m * ldc + n];
        if (act == 1) v = silu_f(v);
        if (Cf)  Cf[(size_t)m * ldc + n]  = v;
        if (Cbf) Cbf[(size_t)m * ldc + n] = (__bf16)v;
      }
    }
  }
}

// ---------------------------------------------------------------------------
// Gate logits: it[b,h,s] = q[m,:]·wi[:,h] + wi_b[h]; ft likewise from k.
// One thread per (row, head).
// ---------------------------------------------------------------------------
__global__ void k_gates(const __bf16* __restrict__ q, const __bf16* __restrict__ k,
                        const float* __restrict__ wi, const float* __restrict__ wib,
                        const float* __restrict__ wf, const float* __restrict__ wfb,
                        float* __restrict__ it, float* __restrict__ ft) {
  int t = blockIdx.x * 256 + threadIdx.x;
  if (t >= 4096 * 8) return;
  int m = t >> 3, h = t & 7;
  const __bf16* qr = q + (size_t)m * 1024;
  const __bf16* kr = k + (size_t)m * 1024;
  float si = 0.0f, sf = 0.0f;
  for (int d = 0; d < 1024; d++) {
    si += (float)qr[d] * wi[d * 8 + h];
    sf += (float)kr[d] * wf[d * 8 + h];
  }
  int b = m >> 11, s = m & 2047;
  size_t o = (size_t)(b * 8 + h) * 2048 + s;
  it[o] = si + wib[h];
  ft[o] = sf + wfb[h];
}

// ---------------------------------------------------------------------------
// Sequential scan per (b,h): cs = cumsum(log_sigmoid(ft)),
// m = cs + cummax(it - cs). 16 lanes, one (b,h) each.
// ---------------------------------------------------------------------------
__global__ void k_scan(const float* __restrict__ it, const float* __restrict__ ft,
                       float* __restrict__ cs, float* __restrict__ mrow) {
  int t = threadIdx.x;
  if (t >= 16) return;
  size_t base = (size_t)t * 2048;
  float c = 0.0f, gm = -1e30f;
  for (int s = 0; s < 2048; s++) {
    float f = ft[base + s];
    float ls = (f >= 0.0f) ? -log1pf(__expf(-f)) : (f - log1pf(__expf(f)));
    c += ls;
    float g = it[base + s] - c;
    gm = fmaxf(gm, g);
    cs[base + s] = c;
    mrow[base + s] = c + gm;
  }
}

// ---------------------------------------------------------------------------
// V transpose for the P·V WMMA (B needs K(=j) contiguous per lane):
// vt[(b,h,d,s)] = vh[(b,s,h,d)].
// ---------------------------------------------------------------------------
__global__ void k_vt(const __bf16* __restrict__ vh, __bf16* __restrict__ vt) {
  int t = blockIdx.x * 256 + threadIdx.x;
  if (t >= 2 * 8 * 128 * 2048) return;
  int s = t & 2047;
  int rest = t >> 11;
  int d = rest & 127; rest >>= 7;
  int h = rest & 7;
  int b = rest >> 3;
  vt[t] = vh[((size_t)(b * 2048 + s)) * 1024 + h * 128 + d];
}

// ---------------------------------------------------------------------------
// mLSTM attention core + GroupNorm. Flash-style over j-tiles of 32.
// D[i,j] = exp((cs[i]-m[i]) + (it[j]-cs[j])), exponent <= 0 for j<=i.
// Grid: (32 i-blocks, 8 heads, 2 batch), 128 threads (4 waves),
// each wave owns 16 query rows. Per-wave LDS regions, no workgroup barriers
// (DS ordering via s_wait_dscnt).
// ---------------------------------------------------------------------------
__global__ void k_attn(const __bf16* __restrict__ qh, const __bf16* __restrict__ kh,
                       const __bf16* __restrict__ vt,
                       const float* __restrict__ cs, const float* __restrict__ it,
                       const float* __restrict__ mrow,
                       const float* __restrict__ gn_w, const float* __restrict__ gn_b,
                       float* __restrict__ Hn) {
  int h = blockIdx.y, b = blockIdx.z;
  int lane = threadIdx.x & 31, w = threadIdx.x >> 5;
  int ib = blockIdx.x * 64 + w * 16;
  int lr = lane & 15, lh = lane >> 4;

  __shared__ alignas(64) __bf16 Pst[4][16][32];
  __shared__ alignas(64) float  Hst[4][16][128];
  __shared__ float Rs[4][16];

  const size_t qoff = (size_t)(b * 2048) * 1024 + (size_t)h * 128;
  const size_t goff = (size_t)(b * 8 + h) * 2048;
  const size_t voff = (size_t)((b * 8 + h) * 128) * 2048;

  // Q fragments: 16 rows x 128 d, A-layout (lane = row, 16 bf16 along K).
  v16bf qf[4];
  for (int kd = 0; kd < 4; kd++)
    qf[kd] = *(const v16bf*)(qh + qoff + (size_t)(ib + lr) * 1024 + kd * 32 + lh * 16);

  // Row constants for this lane's 8 accumulator rows (i = ib + r + 8*lh).
  float csi[8], mi[8];
  for (int r = 0; r < 8; r++) {
    int i = ib + r + 8 * lh;
    csi[r] = cs[goff + i];
    mi[r]  = mrow[goff + i];
  }

  v8f acc[8] = {};
  float rsum = 0.0f;              // lane L accumulates rowsum of row (L&15)
  const float invtau = 0.03125f;  // 1/sqrt(1024)

  for (int jt = 0; jt <= ib + 15; jt += 32) {
    // ---- S = Q K^T for two 16-column subtiles, scale into P, stage in LDS.
    for (int js = 0; js < 2; js++) {
      int j0 = jt + js * 16;
      v8f s = {};
      for (int kd = 0; kd < 4; kd++) {
        v16bf kf = *(const v16bf*)(kh + qoff + (size_t)(j0 + lr) * 1024 + kd * 32 + lh * 16);
        s = WMMA_BF16(qf[kd], kf, s);
      }
      int jcol = j0 + lr;                // this lane's output column
      float cj = cs[goff + jcol];
      float tj = it[goff + jcol];
      for (int r = 0; r < 8; r++) {
        int i = ib + r + 8 * lh;
        float p = 0.0f;
        if (jcol <= i)
          p = s[r] * invtau * __expf(csi[r] - cj + tj - mi[r]);
        Pst[w][r + 8 * lh][js * 16 + lr] = (__bf16)p;
      }
    }
    asm volatile("s_wait_dscnt 0" ::: "memory");  // wave-local LDS ordering

    // Row-sum of Ctilde tile (for maxit). Both half-waves compute same rows.
    {
      float rs = 0.0f;
      for (int j = 0; j < 32; j++) rs += (float)Pst[w][lr][j];
      rsum += rs;
    }

    // Re-load P as an A-fragment (lane = row, 16 bf16 contiguous along j).
    v16bf pf = *(const v16bf*)(&Pst[w][lr][lh * 16]);

    // H += P * V over hd via 8 N-subtiles, K = 32 (this j-tile).
    for (int f = 0; f < 8; f++) {
      int d = f * 16 + lr;
      v16bf vf = *(const v16bf*)(vt + voff + (size_t)d * 2048 + jt + lh * 16);
      acc[f] = WMMA_BF16(pf, vf, acc[f]);
    }
  }

  Rs[w][lr] = rsum;  // lanes 0-15 and 16-31 write identical values
  asm volatile("s_wait_dscnt 0" ::: "memory");

  // maxit scaling, dump H to LDS for per-row GroupNorm.
  float scale[8];
  for (int r = 0; r < 8; r++) {
    int idx = r + 8 * lh;
    float mx = fmaxf(fabsf(Rs[w][idx]), __expf(-mi[r]));
    scale[r] = 1.0f / (mx + 1e-8f);
  }
  for (int f = 0; f < 8; f++)
    for (int r = 0; r < 8; r++)
      Hst[w][r + 8 * lh][f * 16 + lr] = acc[f][r] * scale[r];
  asm volatile("s_wait_dscnt 0" ::: "memory");

  // GroupNorm over HEAD_DIM=128 per row, lanes 0-15 each own one row.
  if (lh == 0) {
    float s1 = 0.0f, s2 = 0.0f;
    for (int d = 0; d < 128; d++) {
      float v = Hst[w][lr][d];
      s1 += v; s2 += v * v;
    }
    float mean = s1 * (1.0f / 128.0f);
    float var  = s2 * (1.0f / 128.0f) - mean * mean;
    float rstd = rsqrtf(var + 1e-5f);
    int i = ib + lr;
    float* outp = Hn + ((size_t)(b * 2048 + i)) * 1024 + h * 128;
    for (int d = 0; d < 128; d++) {
      float v = (Hst[w][lr][d] - mean) * rstd;
      outp[d] = v * gn_w[h * 128 + d] + gn_b[h * 128 + d];
    }
  }
}

// ---------------------------------------------------------------------------
// E = (Hn + skip*qk) * bgate  -> bf16 for final GEMM.
// ---------------------------------------------------------------------------
__global__ void k_combine(const float* __restrict__ Hn, const float* __restrict__ skip,
                          const float* __restrict__ qk, const float* __restrict__ bgate,
                          __bf16* __restrict__ E) {
  int t = blockIdx.x * 256 + threadIdx.x;
  if (t >= 4096 * 1024) return;
  float v = (Hn[t] + skip[t & 1023] * qk[t]) * bgate[t];
  E[t] = (__bf16)v;
}

// ---------------------------------------------------------------------------
// Host-side orchestration.
// ---------------------------------------------------------------------------
extern "C" void kernel_launch(void* const* d_in, const int* in_sizes, int n_in,
                              void* d_out, int out_size, void* d_ws, size_t ws_size,
                              hipStream_t stream) {
  const float* x      = (const float*)d_in[0];
  const float* ln_w   = (const float*)d_in[1];
  const float* ln_b   = (const float*)d_in[2];
  const float* mlp1_w = (const float*)d_in[3];
  const float* mlp1_b = (const float*)d_in[4];
  const float* mlp2_w = (const float*)d_in[5];
  const float* mlp2_b = (const float*)d_in[6];
  const float* conv_w = (const float*)d_in[7];
  const float* conv_b = (const float*)d_in[8];
  const float* bq_w   = (const float*)d_in[9];
  const float* bk_w   = (const float*)d_in[10];
  const float* bv_w   = (const float*)d_in[11];
  const float* wq_w   = (const float*)d_in[12];
  const float* wq_b   = (const float*)d_in[13];
  const float* wk_w   = (const float*)d_in[14];
  const float* wk_b   = (const float*)d_in[15];
  const float* wv_w   = (const float*)d_in[16];
  const float* wv_b   = (const float*)d_in[17];
  const float* wi_w   = (const float*)d_in[18];
  const float* wi_b   = (const float*)d_in[19];
  const float* wf_w   = (const float*)d_in[20];
  const float* wf_b   = (const float*)d_in[21];
  const float* gn_w   = (const float*)d_in[22];
  const float* gn_b   = (const float*)d_in[23];
  const float* skip   = (const float*)d_in[24];
  const float* fin_w  = (const float*)d_in[25];
  const float* fin_b  = (const float*)d_in[26];
  float* out = (float*)d_out;

  char* p = (char*)d_ws;
  auto alloc = [&](size_t bytes) -> void* {
    void* r = (void*)p;
    p += (bytes + 255) & ~(size_t)255;
    return r;
  };

  const size_t M = 4096;  // 2 * 2048 rows
  __bf16* xn     = (__bf16*)alloc(M * 512 * 2);
  __bf16* a_pad  = (__bf16*)alloc((size_t)2 * 2051 * 1024 * 2);
  __bf16* mlp1_t = (__bf16*)alloc((size_t)1024 * 512 * 2);
  __bf16* mlp2_t = (__bf16*)alloc((size_t)1024 * 512 * 2);
  __bf16* conv_t = (__bf16*)alloc((size_t)4 * 1024 * 1024 * 2);
  __bf16* bq_t   = (__bf16*)alloc((size_t)4 * 256 * 256 * 2);
  __bf16* bk_t   = (__bf16*)alloc((size_t)4 * 256 * 256 * 2);
  __bf16* bv_t   = (__bf16*)alloc((size_t)4 * 256 * 256 * 2);
  __bf16* wq_t   = (__bf16*)alloc((size_t)1024 * 1024 * 2);
  __bf16* wk_t   = (__bf16*)alloc((size_t)1024 * 1024 * 2);
  __bf16* wv_t   = (__bf16*)alloc((size_t)1024 * 1024 * 2);
  __bf16* fin_t  = (__bf16*)alloc((size_t)512 * 1024 * 2);
  float*  bgate  = (float*) alloc(M * 1024 * 4);
  float*  qk_f   = (float*) alloc(M * 1024 * 4);
  __bf16* qk_b   = (__bf16*)alloc(M * 1024 * 2);
  __bf16* qb     = (__bf16*)alloc(M * 1024 * 2);
  __bf16* kb     = (__bf16*)alloc(M * 1024 * 2);
  __bf16* vb     = (__bf16*)alloc(M * 1024 * 2);
  __bf16* qhb    = (__bf16*)alloc(M * 1024 * 2);
  __bf16* khb    = (__bf16*)alloc(M * 1024 * 2);
  __bf16* vhb    = (__bf16*)alloc(M * 1024 * 2);
  __bf16* vtb    = (__bf16*)alloc(M * 1024 * 2);
  float*  itb    = (float*) alloc((size_t)2 * 8 * 2048 * 4);
  float*  ftb    = (float*) alloc((size_t)2 * 8 * 2048 * 4);
  float*  csb    = (float*) alloc((size_t)2 * 8 * 2048 * 4);
  float*  mb     = (float*) alloc((size_t)2 * 8 * 2048 * 4);
  float*  Hn     = (float*) alloc(M * 1024 * 4);
  __bf16* Eb     = (__bf16*)alloc(M * 1024 * 2);
  (void)ws_size; (void)in_sizes; (void)n_in; (void)out_size;

  // 1. pad rows + layernorm
  k_zero_pad<<<24, 256, 0, stream>>>(a_pad);
  k_layernorm<<<M, 256, 0, stream>>>(x, ln_w, ln_b, xn);

  // 2. weight transpose/convert
  auto wtr = [&](const float* src, __bf16* dst, int K, int N, int sk, int sn) {
    k_w2bf16t<<<(K * N + 255) / 256, 256, 0, stream>>>(src, dst, K, N, sk, sn);
  };
  wtr(mlp1_w, mlp1_t, 512, 1024, 1024, 1);
  wtr(mlp2_w, mlp2_t, 512, 1024, 1024, 1);
  for (int kc = 0; kc < 4; kc++)  // conv_w[o,i,kc] -> Bt_kc[o][i]
    wtr(conv_w + kc, conv_t + (size_t)kc * 1024 * 1024, 1024, 1024, 4, 4096);
  for (int g = 0; g < 4; g++) {
    wtr(bq_w + (size_t)g * 65536, bq_t + (size_t)g * 65536, 256, 256, 256, 1);
    wtr(bk_w + (size_t)g * 65536, bk_t + (size_t)g * 65536, 256, 256, 256, 1);
    wtr(bv_w + (size_t)g * 65536, bv_t + (size_t)g * 65536, 256, 256, 256, 1);
  }
  wtr(wq_w, wq_t, 1024, 1024, 1024, 1);
  wtr(wk_w, wk_t, 1024, 1024, 1024, 1);
  wtr(wv_w, wv_t, 1024, 1024, 1024, 1);
  wtr(fin_w, fin_t, 1024, 512, 512, 1);

  dim3 blk(256);
  auto gemm = [&](const __bf16* A, int lda, const __bf16* Bt, const float* bias,
                  const float* addsrc, float* Cf, __bf16* Cbf, int ldc,
                  int Mrows, int N, int K, int ksteps, int act) {
    dim3 grid(N / 128, Mrows / 128);
    k_gemm<<<grid, blk, 0, stream>>>(A, lda, Bt, bias, addsrc, Cf, Cbf, ldc,
                                     K, N, ksteps, act);
  };

  // 3. a = xn @ mlp1 + b  (bf16, into padded buffer per batch)
  for (int b = 0; b < 2; b++)
    gemm(xn + (size_t)b * 2048 * 512, 512, mlp1_t, mlp1_b, nullptr,
         nullptr, a_pad + ((size_t)b * 2051 + 3) * 1024, 1024, 2048, 1024, 512, 1, 0);
  // 4. bgate = silu(xn @ mlp2 + b)
  gemm(xn, 512, mlp2_t, mlp2_b, nullptr, bgate, nullptr, 1024, 4096, 1024, 512, 1, 1);

  // 5. qk = silu(causal_conv(a) + conv_b): 4 shifted GEMMs accumulated
  for (int b = 0; b < 2; b++)
    gemm(a_pad + (size_t)b * 2051 * 1024, 1024, conv_t, conv_b, nullptr,
         qk_f + (size_t)b * 2048 * 1024, qk_b + (size_t)b * 2048 * 1024, 1024,
         2048, 1024, 1024, 4, 1);

  // 6. block-diagonal q/k (from qk), v (from a)
  for (int g = 0; g < 4; g++) {
    gemm(qk_b + g * 256, 1024, bq_t + (size_t)g * 65536, nullptr, nullptr,
         nullptr, qb + g * 256, 1024, 4096, 256, 256, 1, 0);
    gemm(qk_b + g * 256, 1024, bk_t + (size_t)g * 65536, nullptr, nullptr,
         nullptr, kb + g * 256, 1024, 4096, 256, 256, 1, 0);
    for (int b = 0; b < 2; b++)
      gemm(a_pad + ((size_t)b * 2051 + 3) * 1024 + g * 256, 1024,
           bv_t + (size_t)g * 65536, nullptr, nullptr,
           nullptr, vb + (size_t)b * 2048 * 1024 + g * 256, 1024, 2048, 256, 256, 1, 0);
  }

  // 7. head projections
  gemm(qb, 1024, wq_t, wq_b, nullptr, nullptr, qhb, 1024, 4096, 1024, 1024, 1, 0);
  gemm(kb, 1024, wk_t, wk_b, nullptr, nullptr, khb, 1024, 4096, 1024, 1024, 1, 0);
  gemm(vb, 1024, wv_t, wv_b, nullptr, nullptr, vhb, 1024, 4096, 1024, 1024, 1, 0);
  k_vt<<<(4096 * 1024) / 256, 256, 0, stream>>>(vhb, vtb);

  // 8. gate logits + scan (cs, m)
  k_gates<<<(4096 * 8) / 256, 256, 0, stream>>>(qb, kb, wi_w, wi_b, wf_w, wf_b, itb, ftb);
  k_scan<<<1, 32, 0, stream>>>(itb, ftb, csb, mb);

  // 9. mLSTM core + GroupNorm
  k_attn<<<dim3(32, 8, 2), dim3(128), 0, stream>>>(qhb, khb, vtb, csb, itb, mb,
                                                   gn_w, gn_b, Hn);

  // 10. combine, final projection + residual
  k_combine<<<(4096 * 1024) / 256, 256, 0, stream>>>(Hn, skip, qk_f, bgate, Eb);
  gemm(Eb, 1024, fin_t, fin_b, x, out, nullptr, 512, 4096, 512, 1024, 1, 0);
}
